// SelfAttention_12128987644299
// MI455X (gfx1250) — matrix operand order repaired and checked by reading, not compiled, not run on hardware
//
#include <hip/hip_runtime.h>
#include <hip/hip_bf16.h>

// ---------------------------------------------------------------------------
// Self-attention (flash-style) + output projection for gfx1250 (MI455X).
// f16 WMMA (v_wmma_f32_16x16x32_f16) with f32 accumulation, wave32.
// Pre-converted f16 operands; async global->LDS staging (ASYNCcnt path).
// ---------------------------------------------------------------------------

typedef __attribute__((ext_vector_type(16))) _Float16 v16h;
typedef __attribute__((ext_vector_type(8)))  _Float16 v8h;
typedef __attribute__((ext_vector_type(8)))  float    v8f;
typedef __attribute__((ext_vector_type(4)))  int      v4i;

#define N_BATCH 2
#define SEQ_L   2048
#define D_MODEL 1024
#define N_HEADS 16
#define D_V     64          // head dim
#define BM      64          // query rows per workgroup (16 per wave)
#define BN      64          // key block per iteration
#define WAVES   4
#define QSCALE  0.03125f    // 1/sqrt(1024), folded into Q

// ---- gfx1250 async global->LDS staging (ASYNCcnt path), with fallback ----
#if defined(__gfx1250__) && \
    __has_builtin(__builtin_amdgcn_global_load_async_to_lds_b128) && \
    __has_builtin(__builtin_amdgcn_s_wait_asynccnt)
#define HAVE_ASYNC 1
#else
#define HAVE_ASYNC 0
#if defined(__HIP_DEVICE_COMPILE__)
#warning "gfx1250 async-to-LDS builtins unavailable; using synchronous staging"
#endif
#endif

// builtin prototype (from hipcc diagnostic): typed v4i pointers,
// addrspace(1) global source, addrspace(3) LDS destination.
typedef __attribute__((address_space(1))) v4i* g_v4i_p;
typedef __attribute__((address_space(3))) v4i* l_v4i_p;

#define AS1P(p) ((g_v4i_p)(unsigned long long)(p))
#define AS3P(p) ((l_v4i_p)(unsigned)(unsigned long long)(p))

__device__ __forceinline__ void cp16_g2l(void* lds_dst, const void* glb_src) {
#if HAVE_ASYNC
    __builtin_amdgcn_global_load_async_to_lds_b128(AS1P(glb_src), AS3P(lds_dst), 0, 0);
#else
    *(v4i*)lds_dst = *(const v4i*)glb_src;
#endif
}
__device__ __forceinline__ void cp_wait() {
#if HAVE_ASYNC
    __builtin_amdgcn_s_wait_asynccnt(0);
#endif
}

// XOR-butterfly within 16-lane halves via ds_swizzle (imm-encoded pattern)
#define SWZ_XOR_F(x, m) \
    __int_as_float(__builtin_amdgcn_ds_swizzle(__float_as_int(x), (((m) << 10) | 0x1f)))

union frag_u { v16h v; v8h h[2]; };

__device__ __forceinline__ v8f wmma_f16(v16h a, v16h b, v8f c) {
    return __builtin_amdgcn_wmma_f32_16x16x32_f16(
        false, a, false, b, (short)0, c, false, false);
}

// ---------------------------------------------------------------------------
// Kernel A: convert K -> Kh [n,h,key,d] f16 and V -> VhT [n,h,d,key] f16
// (transpose through LDS for coalescing). grid = (L/64, H, N), block = 256.
// ---------------------------------------------------------------------------
__global__ __launch_bounds__(256)
void convert_kv_kernel(const float* __restrict__ K,
                       const float* __restrict__ V,
                       _Float16*    __restrict__ Kh,
                       _Float16*    __restrict__ VhT) {
    __shared__ _Float16 tile[64][72];   // 72-stride dodges bank conflicts

    const int tid = threadIdx.x;
    const int kb  = blockIdx.x * 64;
    const int h   = blockIdx.y;
    const int nb  = blockIdx.z;

    const int idx = tid * 16;           // 256 thr * 16 = 4096 = 64x64
    const int r   = idx >> 6;           // key within tile
    const int c   = idx & 63;           // dim (0,16,32,48)

    const size_t gsrc = ((size_t)nb * SEQ_L + (size_t)(kb + r)) * D_MODEL + h * D_V + c;
    const size_t kdst = ((size_t)(nb * N_HEADS + h) * SEQ_L + (kb + r)) * D_V + c;
    #pragma unroll
    for (int i = 0; i < 16; ++i) {
        Kh[kdst + i]    = (_Float16)K[gsrc + i];
        tile[r][c + i]  = (_Float16)V[gsrc + i];
    }
    __syncthreads();

    // write transposed: row = dim, contiguous over keys
    const int od   = idx >> 6;          // dim
    const int okey = idx & 63;          // key (0,16,32,48)
    const size_t vdst =
        ((size_t)(nb * N_HEADS + h) * D_V + od) * SEQ_L + kb + okey;
    #pragma unroll
    for (int i = 0; i < 16; ++i)
        VhT[vdst + i] = tile[okey + i][od];
}

// ---------------------------------------------------------------------------
// Kernel B: convert fc_w (f32 [out,in]) -> Wh (f16). grid = 1024, block = 256.
// ---------------------------------------------------------------------------
__global__ __launch_bounds__(256)
void convert_w_kernel(const float* __restrict__ W, _Float16* __restrict__ Wh) {
    const size_t i = ((size_t)blockIdx.x * 256 + threadIdx.x) * 4;
    #pragma unroll
    for (int j = 0; j < 4; ++j) Wh[i + j] = (_Float16)W[i + j];
}

// ---------------------------------------------------------------------------
// Kernel C: flash attention for one (batch, head, 64-query tile).
// grid = (L/BM, H, N), block = 128 (4 wave32s).
// Per key block of 64: 8 QK^T WMMAs + 2 rowsum WMMAs + 8 PV WMMAs.
// ---------------------------------------------------------------------------
__global__ __launch_bounds__(128)
void attn_fwd_kernel(const float*    __restrict__ Q,
                     const _Float16* __restrict__ Kh,
                     const _Float16* __restrict__ VhT,
                     const int*      __restrict__ maskp,
                     _Float16*       __restrict__ O) {
    __shared__ alignas(32) _Float16 sK[BN * D_V];         // [key][d]   8 KB
    __shared__ alignas(32) _Float16 sV[D_V * BN];         // [d][key]   8 KB
    __shared__ alignas(32) _Float16 sP[WAVES * 16 * BN];  // per-wave P 8 KB

    const int tid   = threadIdx.x;
    const int wv    = tid >> 5;
    const int lane  = tid & 31;
    const int lhalf = lane >> 4;
    const int l16   = lane & 15;

    const int q0 = blockIdx.x * BM;
    const int h  = blockIdx.y;
    const int nb = blockIdx.z;
    const int nh = nb * N_HEADS + h;

    // ---- Q A-fragments (16 rows x 64 dims -> two 16x32 fragments) ----
    const float* qbase =
        Q + ((size_t)nb * SEQ_L + (size_t)(q0 + wv * 16 + l16)) * D_MODEL + h * D_V;
    v16h qa[2];
    #pragma unroll
    for (int f = 0; f < 2; ++f) {
        const int ks = 32 * f;
        const int b  = lhalf * 8;
        #pragma unroll
        for (int i = 0; i < 8; ++i) {
            qa[f][i]     = (_Float16)(qbase[ks + b + i]      * QSCALE);
            qa[f][8 + i] = (_Float16)(qbase[ks + b + 16 + i] * QSCALE);
        }
    }

    // all-ones 32x16 B fragment for row-sum-by-WMMA
    v16h ones;
    #pragma unroll
    for (int i = 0; i < 16; ++i) ones[i] = (_Float16)1.0f;

    v8f acc[4] = {v8f{}, v8f{}, v8f{}, v8f{}};
    float rowmax[8], rowsum[8], fac[8];
    #pragma unroll
    for (int j = 0; j < 8; ++j) { rowmax[j] = -3.0e38f; rowsum[j] = 0.0f; }

    const float NEGBIG = -1.0e30f;

    const _Float16* ktile_base = Kh + (size_t)nh * SEQ_L * D_V;           // row-major
    const _Float16* vtile_base = VhT + (size_t)nh * D_V * SEQ_L;          // [d][key]

    for (int kb = 0; kb < SEQ_L; kb += BN) {
        __syncthreads();
        // ---- async-stage K tile (contiguous 8 KB) and V^T tile (64x128B) ----
        {
            const char* kg = (const char*)(ktile_base + (size_t)kb * D_V);
            #pragma unroll
            for (int i = 0; i < 4; ++i) {
                const int off = (tid + 128 * i) * 16;    // bytes
                cp16_g2l((char*)sK + off, kg + off);
            }
            #pragma unroll
            for (int i = 0; i < 4; ++i) {
                const int c   = tid + 128 * i;           // 512 chunks of 16B
                const int row = c >> 3;                  // d
                const int sub = c & 7;
                const char* vg = (const char*)(vtile_base + (size_t)row * SEQ_L + kb)
                                 + sub * 16;
                cp16_g2l((char*)sV + row * 128 + sub * 16, vg);
            }
        }
        cp_wait();
        __syncthreads();

        // per-lane key-padding mask for the four 16-wide column tiles
        int mk[4];
        #pragma unroll
        for (int nt = 0; nt < 4; ++nt)
            mk[nt] = maskp[nb * SEQ_L + kb + 16 * nt + l16];

        // ---- S = Q * K^T (16 x 64 tile) ----
        v8f s[4] = {v8f{}, v8f{}, v8f{}, v8f{}};
        #pragma unroll
        for (int nt = 0; nt < 4; ++nt) {
            #pragma unroll
            for (int ks2 = 0; ks2 < 2; ++ks2) {
                const v8h* kp = reinterpret_cast<const v8h*>(
                    &sK[(16 * nt + l16) * D_V + 32 * ks2 + lhalf * 8]);
                frag_u ku; ku.h[0] = kp[0]; ku.h[1] = kp[2];
                const v16h kf = ku.v;
                s[nt] = wmma_f16(qa[ks2], kf, s[nt]);
            }
        }

        // ---- online softmax: max + exp + P store (rowsum comes via WMMA) ----
        #pragma unroll
        for (int j = 0; j < 8; ++j) {
            float a0 = mk[0] ? s[0][j] : NEGBIG;
            float a1 = mk[1] ? s[1][j] : NEGBIG;
            float a2 = mk[2] ? s[2][j] : NEGBIG;
            float a3 = mk[3] ? s[3][j] : NEGBIG;
            float m = fmaxf(fmaxf(a0, a1), fmaxf(a2, a3));
            m = fmaxf(m, SWZ_XOR_F(m, 1));
            m = fmaxf(m, SWZ_XOR_F(m, 2));
            m = fmaxf(m, SWZ_XOR_F(m, 4));
            m = fmaxf(m, SWZ_XOR_F(m, 8));
            const float nm = fmaxf(rowmax[j], m);
            fac[j]    = __expf(rowmax[j] - nm);
            rowmax[j] = nm;
            const float e0 = __expf(a0 - nm);
            const float e1 = __expf(a1 - nm);
            const float e2 = __expf(a2 - nm);
            const float e3 = __expf(a3 - nm);
            const int prow = wv * 16 * BN + (j + lhalf * 8) * BN;
            sP[prow + l16]      = (_Float16)e0;
            sP[prow + 16 + l16] = (_Float16)e1;
            sP[prow + 32 + l16] = (_Float16)e2;
            sP[prow + 48 + l16] = (_Float16)e3;
            #pragma unroll
            for (int t = 0; t < 4; ++t) acc[t][j] *= fac[j];
        }

        asm volatile("s_wait_dscnt 0" ::: "memory");

        // ---- P as two 16x32 A-fragments ----
        frag_u pu[2];
        #pragma unroll
        for (int kk = 0; kk < 2; ++kk) {
            const v8h* pp = reinterpret_cast<const v8h*>(
                &sP[wv * 16 * BN + l16 * BN + kk * 32 + lhalf * 8]);
            pu[kk].h[0] = pp[0];   // K = base + 0..7
            pu[kk].h[1] = pp[2];   // K = base + 16..23
        }

        // ---- row sums of P via ones-matrix WMMA (replicated per column) ----
        v8f rs = v8f{};
        rs = wmma_f16(pu[0].v, ones, rs);
        rs = wmma_f16(pu[1].v, ones, rs);
        #pragma unroll
        for (int j = 0; j < 8; ++j)
            rowsum[j] = rowsum[j] * fac[j] + rs[j];

        // ---- acc += P * V ----
        #pragma unroll
        for (int t = 0; t < 4; ++t) {
            #pragma unroll
            for (int kk = 0; kk < 2; ++kk) {
                const v8h* vp = reinterpret_cast<const v8h*>(
                    &sV[(16 * t + l16) * BN + kk * 32 + lhalf * 8]);
                frag_u vu; vu.h[0] = vp[0]; vu.h[1] = vp[2];
                const v16h vf = vu.v;
                acc[t] = wmma_f16(pu[kk].v, vf, acc[t]);
            }
        }
    }

    // ---- normalize and store O (f16, [n*L + q][h*64 + d]) ----
    #pragma unroll
    for (int t = 0; t < 4; ++t) {
        #pragma unroll
        for (int j = 0; j < 8; ++j) {
            const int row = q0 + wv * 16 + j + lhalf * 8;
            const int col = h * D_V + 16 * t + l16;
            O[((size_t)nb * SEQ_L + row) * D_MODEL + col] =
                (_Float16)(acc[t][j] / rowsum[j]);
        }
    }
}

// ---------------------------------------------------------------------------
// Kernel D: out = O @ Wh^T + fc_b.  O: [4096 x 1024] f16, Wh: [out,in] f16.
// grid = (NL/64, D/64), block = 128. Each wave computes a 16 x 64 tile.
// ---------------------------------------------------------------------------
__global__ __launch_bounds__(128)
void proj_kernel(const _Float16* __restrict__ O,
                 const _Float16* __restrict__ Wh,
                 const float*    __restrict__ Bv,
                 float*          __restrict__ out) {
    __shared__ alignas(32) _Float16 sW[64 * 32];   // [col][k] 4 KB

    const int tid   = threadIdx.x;
    const int wv    = tid >> 5;
    const int lane  = tid & 31;
    const int lhalf = lane >> 4;
    const int l16   = lane & 15;

    const int r0 = blockIdx.x * 64;
    const int c0 = blockIdx.y * 64;

    v8f acc[4] = {v8f{}, v8f{}, v8f{}, v8f{}};

    for (int k0 = 0; k0 < D_MODEL; k0 += 32) {
        __syncthreads();
        // async-stage W tile: 64 cols x 32 k (row = 64 B)
        #pragma unroll
        for (int i = 0; i < 2; ++i) {
            const int c   = tid + 128 * i;   // 256 chunks of 16B
            const int row = c >> 2;          // output column
            const int sub = c & 3;
            const char* wg = (const char*)(Wh + (size_t)(c0 + row) * D_MODEL + k0)
                             + sub * 16;
            cp16_g2l((char*)sW + row * 64 + sub * 16, wg);
        }
        cp_wait();
        __syncthreads();

        // A-fragment straight from global f16 O
        frag_u au;
        {
            const _Float16* og =
                O + (size_t)(r0 + wv * 16 + l16) * D_MODEL + k0 + lhalf * 8;
            au.h[0] = *reinterpret_cast<const v8h*>(og);
            au.h[1] = *reinterpret_cast<const v8h*>(og + 16);
        }

        #pragma unroll
        for (int t = 0; t < 4; ++t) {
            const v8h* wp = reinterpret_cast<const v8h*>(
                &sW[(16 * t + l16) * 32 + lhalf * 8]);
            frag_u wu; wu.h[0] = wp[0]; wu.h[1] = wp[2];
            const v16h bf = wu.v;
            acc[t] = wmma_f16(au.v, bf, acc[t]);
        }
    }

    #pragma unroll
    for (int t = 0; t < 4; ++t) {
        const int col  = c0 + 16 * t + l16;
        const float bb = Bv[col];
        #pragma unroll
        for (int j = 0; j < 8; ++j) {
            const int row = r0 + wv * 16 + j + lhalf * 8;
            out[(size_t)row * D_MODEL + col] = acc[t][j] + bb;
        }
    }
}

// ---------------------------------------------------------------------------
extern "C" void kernel_launch(void* const* d_in, const int* in_sizes, int n_in,
                              void* d_out, int out_size, void* d_ws, size_t ws_size,
                              hipStream_t stream) {
    const float* q    = (const float*)d_in[0];
    const float* k    = (const float*)d_in[1];
    const float* v    = (const float*)d_in[2];
    const int*   mask = (const int*)  d_in[3];
    const float* fcw  = (const float*)d_in[4];
    const float* fcb  = (const float*)d_in[5];
    float*       out  = (float*)d_out;

    // workspace layout (f16):
    //   O   : 4096*1024            =  8 MB
    //   Kh  : 2*16*2048*64         =  8 MB
    //   VhT : 2*16*64*2048         =  8 MB
    //   Wh  : 1024*1024            =  2 MB
    char* ws = (char*)d_ws;
    _Float16* O   = (_Float16*)(ws);
    _Float16* Kh  = (_Float16*)(ws + (size_t)8  * 1024 * 1024);
    _Float16* VhT = (_Float16*)(ws + (size_t)16 * 1024 * 1024);
    _Float16* Wh  = (_Float16*)(ws + (size_t)24 * 1024 * 1024);

    convert_kv_kernel<<<dim3(SEQ_L / 64, N_HEADS, N_BATCH), 256, 0, stream>>>(
        k, v, Kh, VhT);
    convert_w_kernel<<<dim3((D_MODEL * D_MODEL) / (256 * 4)), 256, 0, stream>>>(
        fcw, Wh);

    attn_fwd_kernel<<<dim3(SEQ_L / BM, N_HEADS, N_BATCH), 128, 0, stream>>>(
        q, Kh, VhT, mask, O);

    proj_kernel<<<dim3((N_BATCH * SEQ_L) / 64, D_MODEL / 64), 128, 0, stream>>>(
        O, Wh, fcb, out);
}
